// RNN_test_49374944035362
// MI455X (gfx1250) — matrix operand order repaired and checked by reading, not compiled, not run on hardware
//
#include <hip/hip_runtime.h>
#include <math.h>

#define BATCH   64
#define TSTEPS  1000
#define CIN     700
#define N1      512
#define IN1     1212
#define K1P     1216          // IN1 padded to 32
#define NK1     38            // K1P/32
#define N2      256
#define IN2     768
#define NK2     24            // IN2/32
#define N1ROWS  2048          // N1*BR
#define N2ROWS  1024          // N2*BR
#define NWG     64
#define R1      32            // layer-1 rows per WG
#define R2      16            // layer-2 rows per WG
#define NU1     8             // layer-1 neurons per WG
#define NU2     4             // layer-2 neurons per WG
#define WGSIZE  256

// LDS layout (bytes)
#define OFF_W1  0             // R1*K1P bf16  = 77824 B
#define OFF_W2  77824         // R2*IN2 bf16  = 24576 B
#define OFF_SCR 102400        // 2048 f32     =  8192 B
#define OFF_D2  110592        // 64*16 f32    =  4096 B
#define OFF_CST 114688        // consts       =   512 B
#define SLDS    115200

typedef __attribute__((ext_vector_type(16))) __bf16   v16bf;
typedef __attribute__((ext_vector_type(8)))  float    v8f;
typedef __attribute__((ext_vector_type(4)))  unsigned u32x4;

// Persistent device state (rewritten every launch -> deterministic)
__device__ __attribute__((aligned(16))) __bf16 g_W1[N1ROWS * K1P];
__device__ __attribute__((aligned(16))) __bf16 g_W2[N2ROWS * IN2];
__device__ __attribute__((aligned(16))) __bf16 g_k1[2][BATCH * K1P]; // [x | s1 | pad]
__device__ __attribute__((aligned(16))) __bf16 g_k2[2][BATCH * IN2]; // [s1 | s2]
__device__ unsigned g_cnt;
__device__ unsigned g_flag;

__device__ __forceinline__ float sigmoidf_(float v) { return 1.0f / (1.0f + __expf(-v)); }

// A-matrix (MxK) bf16 fragment: lanes 0-15 hold K[0..8)&K[16..24), lanes 16-31 K[8..16)&K[24..32)
__device__ __forceinline__ v16bf frag_A(const __bf16* rowp, int kb, int hiA) {
  union { u32x4 q[2]; v16bf v; } u;
  const __bf16* p = rowp + kb * 32 + hiA;
  u.q[0] = *(const u32x4*)(p);
  u.q[1] = *(const u32x4*)(p + 16);
  return u.v;
}
// B-matrix (KxN) bf16 fragment from row-major W[N][K]: lanes 0-15 hold K[0..8)&K[16..24), lanes 16-31 K[8..16)&K[24..32)
__device__ __forceinline__ v16bf frag_B(const __bf16* rowp, int kb, int hiB) {
  union { u32x4 q[2]; v16bf v; } u;
  const __bf16* p = rowp + kb * 32 + (hiB >> 1);
  u.q[0] = *(const u32x4*)(p);
  u.q[1] = *(const u32x4*)(p + 16);
  return u.v;
}

// Device-wide sense barrier (generation counter), agent-scope atomics + s_sleep spin
__device__ __forceinline__ void gridBarrier(unsigned gen) {
  __threadfence();
  __syncthreads();
  if (threadIdx.x == 0) {
    unsigned target = (gen + 1u) * (unsigned)NWG;
    unsigned old = __hip_atomic_fetch_add(&g_cnt, 1u, __ATOMIC_ACQ_REL, __HIP_MEMORY_SCOPE_AGENT);
    if (old + 1u == target) {
      __hip_atomic_store(&g_flag, gen + 1u, __ATOMIC_RELEASE, __HIP_MEMORY_SCOPE_AGENT);
    } else {
      while (__hip_atomic_load(&g_flag, __ATOMIC_ACQUIRE, __HIP_MEMORY_SCOPE_AGENT) < gen + 1u) {
        __builtin_amdgcn_s_sleep(2);
      }
    }
  }
  __syncthreads();
  __threadfence();
}

__global__ void __launch_bounds__(WGSIZE)
snn_init() {
  size_t i = (size_t)blockIdx.x * blockDim.x + threadIdx.x;
  const size_t n1 = (size_t)2 * BATCH * K1P;   // 155648
  const size_t n2 = (size_t)2 * BATCH * IN2;   //  98304
  if (i < n1)            (&g_k1[0][0])[i] = (__bf16)0.0f;
  else if (i < n1 + n2)  (&g_k2[0][0])[i - n1] = (__bf16)0.0f;
  if (i == 0) { g_cnt = 0u; g_flag = 0u; }
}

__global__ void __launch_bounds__(WGSIZE)
snn_prep(const float* __restrict__ W1, const float* __restrict__ mask1,
         const float* __restrict__ W2, const float* __restrict__ mask2) {
  size_t i = (size_t)blockIdx.x * blockDim.x + threadIdx.x;
  const size_t t1 = (size_t)N1ROWS * K1P;   // padded W1
  if (i < t1) {
    int row = (int)(i / K1P), k = (int)(i % K1P);
    float v = (k < IN1) ? W1[(size_t)row * IN1 + k] * mask1[(size_t)row * IN1 + k] : 0.0f;
    g_W1[i] = (__bf16)v;
  } else {
    size_t j = i - t1;
    if (j < (size_t)N2ROWS * IN2) g_W2[j] = (__bf16)(W2[j] * mask2[j]);
  }
}

__global__ void __launch_bounds__(WGSIZE, 1)
snn_persist(const float* __restrict__ x,
            const float* __restrict__ b1, const float* __restrict__ tau_m1, const float* __restrict__ tau_n1,
            const float* __restrict__ b2, const float* __restrict__ tau_m2, const float* __restrict__ tau_n2,
            const float* __restrict__ W3, const float* __restrict__ b3, const float* __restrict__ tau_m3,
            float* __restrict__ out)
{
  extern __shared__ __align__(16) char smem[];
  __bf16* sW1 = (__bf16*)(smem + OFF_W1);
  __bf16* sW2 = (__bf16*)(smem + OFF_W2);
  float*  scr = (float*)(smem + OFF_SCR);
  float*  sd2 = (float*)(smem + OFF_D2);
  float*  cst = (float*)(smem + OFF_CST);
  float* c_beta1  = cst +   0;  // 32
  float* c_b1     = cst +  32;  // 32
  float* c_alpha1 = cst +  64;  // 8
  float* c_beta2  = cst +  72;  // 16
  float* c_b2     = cst +  88;  // 16
  float* c_alpha2 = cst + 104;  // 4

  const int wg   = blockIdx.x;
  const int tid  = threadIdx.x;
  const int lane = tid & 31;
  const int wave = tid >> 5;
  const int mt   = wave & 3;        // M (batch) tile
  const int nt   = wave >> 2;       // layer-1 N tile / layer-2 K half
  const int l15  = lane & 15;
  const int hiA  = (lane >> 4) * 8;
  const int hiB  = (lane >> 4) * 16;
  const int mBase = mt * 16 + l15;  // batch row for A fragments
  const int rbase1 = wg * R1, rbase2 = wg * R2;
  const int nbase1 = wg * NU1, nbase2 = wg * NU2;

  // ---- prologue: per-WG constants, LDS-resident weights, x_0 conversion ----
  if (tid < R1)  { c_beta1[tid] = sigmoidf_(tau_n1[rbase1 + tid]); c_b1[tid] = b1[rbase1 + tid]; }
  if (tid < NU1) { c_alpha1[tid] = sigmoidf_(tau_m1[nbase1 + tid]); }
  if (tid < R2)  { c_beta2[tid] = sigmoidf_(tau_n2[rbase2 + tid]); c_b2[tid] = b2[rbase2 + tid]; }
  if (tid < NU2) { c_alpha2[tid] = sigmoidf_(tau_m2[nbase2 + tid]); }
  for (int i = tid; i < BATCH * R2; i += WGSIZE) sd2[i] = 0.0f;
  {
    const u32x4* s1p = (const u32x4*)(g_W1 + (size_t)rbase1 * K1P);
    u32x4* d1p = (u32x4*)sW1;
    for (int i = tid; i < (R1 * K1P) / 8; i += WGSIZE) d1p[i] = s1p[i];
    const u32x4* s2p = (const u32x4*)(g_W2 + (size_t)rbase2 * IN2);
    u32x4* d2p = (u32x4*)sW2;
    for (int i = tid; i < (R2 * IN2) / 8; i += WGSIZE) d2p[i] = s2p[i];
  }
  // each WG converts batch-row `wg` of x at t=0 into k1[0]
  for (int i = tid; i < CIN; i += WGSIZE)
    g_k1[0][(size_t)wg * K1P + i] = (__bf16)x[(size_t)wg * TSTEPS * CIN + i];

  const float a3  = sigmoidf_(tau_m3[tid & 1]);
  const float bb3 = b3[tid & 1];

  // register-resident state for the whole sequence
  v8f d1 = {0.f,0.f,0.f,0.f,0.f,0.f,0.f,0.f};
  float mem1r[2] = {0.f, 0.f};
  float s1r[2]   = {0.f, 0.f};
  float mem2 = 0.f, s2v = 0.f;
  float m3 = 0.f, m3acc = 0.f;

  unsigned gen = 0;
  gridBarrier(gen); gen++;

  for (int t = 0; t < TSTEPS; ++t) {
    const int p = t & 1;
    // ---------------- layer 1 GEMM: ff1 = k1 @ Wm1^T ----------------
    v8f acc = {0.f,0.f,0.f,0.f,0.f,0.f,0.f,0.f};
    {
      const __bf16* arow = g_k1[p] + (size_t)mBase * K1P;
      const __bf16* brow = sW1 + (size_t)(nt * 16 + l15) * K1P;
      for (int kb = 0; kb < NK1; ++kb) {
        v16bf a = frag_A(arow, kb, hiA);
        v16bf b = frag_B(brow, kb, hiB);
        acc = __builtin_amdgcn_wmma_f32_16x16x32_bf16(false, a, false, b, (short)0, acc, false, false);
      }
    }
    // dendritic state update (per-lane column constants), stash to LDS for branch sums
    {
      int ncol = nt * 16 + l15;
      float be = c_beta1[ncol], bi = c_b1[ncol];
#pragma unroll
      for (int r = 0; r < 8; ++r) {
        float dn = be * d1[r] + (1.0f - be) * (acc[r] + bi);
        d1[r] = dn;
        int m = mt * 16 + r + ((lane >> 4) << 3);
        scr[m * R1 + ncol] = dn;
      }
    }
    __syncthreads();
    // layer-1 membrane + spike (2 (batch,neuron) items per thread)
#pragma unroll
    for (int q = 0; q < 2; ++q) {
      int idx = 2 * tid + q;
      int m = idx >> 3, j = idx & 7;
      const float* row = scr + m * R1 + 4 * j;
      float l = row[0] + row[1] + row[2] + row[3];
      float al = c_alpha1[j];
      float nm = mem1r[q] * al + (1.0f - al) * l - s1r[q];   // soft reset, VTH=1
      float ns = (nm > 1.0f) ? 1.0f : 0.0f;
      mem1r[q] = nm; s1r[q] = ns;
      __bf16 sb = (__bf16)ns;
      g_k1[1 - p][(size_t)m * K1P + CIN + nbase1 + j] = sb;  // recurrent input for t+1
      g_k2[p][(size_t)m * IN2 + nbase1 + j]           = sb;  // layer-2 input for t
    }
    // prefetch x_{t+1} into opposite k1 buffer (safe: only read after next barrier pair)
    if (t + 1 < TSTEPS) {
      const float* xs = x + ((size_t)wg * TSTEPS + (t + 1)) * CIN;
      __bf16* kd = g_k1[1 - p] + (size_t)wg * K1P;
      for (int i = tid; i < CIN; i += WGSIZE) kd[i] = (__bf16)xs[i];
    }
    gridBarrier(gen); gen++;   // all s1_t visible

    // ---------------- layer 2 GEMM: ff2 = k2 @ Wm2^T (K split over wave halves) --------
    v8f acc2 = {0.f,0.f,0.f,0.f,0.f,0.f,0.f,0.f};
    {
      const __bf16* arow = g_k2[p] + (size_t)mBase * IN2;
      const __bf16* brow = sW2 + (size_t)l15 * IN2;
      const int kb0 = nt * (NK2 / 2);
      for (int kb = kb0; kb < kb0 + NK2 / 2; ++kb) {
        v16bf a = frag_A(arow, kb, hiA);
        v16bf b = frag_B(brow, kb, hiB);
        acc2 = __builtin_amdgcn_wmma_f32_16x16x32_bf16(false, a, false, b, (short)0, acc2, false, false);
      }
    }
    {
      int ncol = l15;
#pragma unroll
      for (int r = 0; r < 8; ++r) {
        int m = mt * 16 + r + ((lane >> 4) << 3);
        scr[(nt * BATCH + m) * R2 + ncol] = acc2[r];   // two K-half partials
      }
    }
    __syncthreads();
    // combine K halves + dendritic state (LDS-resident d2), 4 items/thread
#pragma unroll
    for (int q = 0; q < 4; ++q) {
      int idx = 4 * tid + q;            // 0..1023 = m*16 + r
      int m = idx >> 4, r = idx & 15;
      float ff = scr[m * R2 + r] + scr[(BATCH + m) * R2 + r] + c_b2[r];
      float be = c_beta2[r];
      sd2[idx] = be * sd2[idx] + (1.0f - be) * ff;
    }
    __syncthreads();
    // layer-2 membrane + spike (1 item/thread)
    {
      int m = tid >> 2, j = tid & 3;
      const float* row = sd2 + m * R2 + 4 * j;
      float l = row[0] + row[1] + row[2] + row[3];
      float al = c_alpha2[j];
      float nm = mem2 * al + (1.0f - al) * l - s2v;
      float ns = (nm > 1.0f) ? 1.0f : 0.0f;
      mem2 = nm; s2v = ns;
      g_k2[1 - p][(size_t)m * IN2 + N1 + nbase2 + j] = (__bf16)ns;
    }
    gridBarrier(gen); gen++;   // all s2_t visible

    // ---------------- readout integrator (WG0, 128 threads) ----------------
    if (wg == 0 && tid < 128) {
      int m = tid >> 1, o = tid & 1;
      const __bf16* sp = g_k2[1 - p] + (size_t)m * IN2 + N1;
      float dot = bb3;
      for (int n = 0; n < N2; ++n) dot += (float)sp[n] * W3[o * N2 + n];
      m3 = m3 * a3 + (1.0f - a3) * dot;
      if (t > 0) m3acc += m3;
    }
  }

  if (wg == 0 && tid < 128) {
    int m = tid >> 1, o = tid & 1;
    out[m * 2 + o] = m3acc / (float)TSTEPS;
  }
}

extern "C" void kernel_launch(void* const* d_in, const int* in_sizes, int n_in,
                              void* d_out, int out_size, void* d_ws, size_t ws_size,
                              hipStream_t stream) {
  (void)in_sizes; (void)n_in; (void)out_size; (void)d_ws; (void)ws_size;
  const float* x    = (const float*)d_in[0];
  const float* W1   = (const float*)d_in[1];
  const float* b1   = (const float*)d_in[2];
  const float* tm1  = (const float*)d_in[3];
  const float* tn1  = (const float*)d_in[4];
  const float* m1   = (const float*)d_in[5];
  const float* W2   = (const float*)d_in[6];
  const float* b2   = (const float*)d_in[7];
  const float* tm2  = (const float*)d_in[8];
  const float* tn2  = (const float*)d_in[9];
  const float* m2   = (const float*)d_in[10];
  const float* W3   = (const float*)d_in[11];
  const float* b3   = (const float*)d_in[12];
  const float* tm3  = (const float*)d_in[13];
  float* out = (float*)d_out;

  // allow >64KB dynamic LDS (CDNA5 WGP has 320KB); idempotent, capture-safe (no stream op)
  hipFuncSetAttribute((const void*)snn_persist, hipFuncAttributeMaxDynamicSharedMemorySize, SLDS);

  {
    size_t total = (size_t)2 * BATCH * K1P + (size_t)2 * BATCH * IN2;
    int blocks = (int)((total + WGSIZE - 1) / WGSIZE);
    snn_init<<<blocks, WGSIZE, 0, stream>>>();
  }
  {
    size_t total = (size_t)N1ROWS * K1P + (size_t)N2ROWS * IN2;
    int blocks = (int)((total + WGSIZE - 1) / WGSIZE);
    snn_prep<<<blocks, WGSIZE, 0, stream>>>(W1, m1, W2, m2);
  }
  snn_persist<<<NWG, WGSIZE, SLDS, stream>>>(x, b1, tm1, tn1, b2, tm2, tn2, W3, b3, tm3, out);
}
